// MemoryGaussianMoELayer_5592047419819
// MI455X (gfx1250) — matrix operand order repaired and checked by reading, not compiled, not run on hardware
//
#include <hip/hip_runtime.h>
#include <math.h>

// ---------------------------------------------------------------------------
// MemoryGaussianMoELayer for MI455X (gfx1250), bf16 WMMA path.
// Problem constants (from reference setup_inputs).
// ---------------------------------------------------------------------------
constexpr int B_ = 8, T_ = 1024, D_ = 512, H_ = 1024, E_ = 8;
constexpr int N_ = B_ * T_;                   // 8192 tokens
constexpr int TM_ = 32;                       // token tile (2x 16-row WMMA tiles)
constexpr float LOG2PI_ = 1.8378770664093453f;

typedef __attribute__((ext_vector_type(16))) __bf16          v16bf;
typedef __attribute__((ext_vector_type(8)))  float           v8f;
typedef __attribute__((ext_vector_type(8)))  unsigned short  ushort8;

union BF16x16 { ushort8 u[2]; v16bf v; };

__device__ __forceinline__ unsigned short f2bf(float f) {
  // round-to-nearest-even fp32 -> bf16
  unsigned int u = __float_as_uint(f);
  unsigned int r = ((u >> 16) & 1u) + 0x7fffu;
  return (unsigned short)((u + r) >> 16);
}

// ---------------------------------------------------------------------------
// Kernel 0: zero y (scatter-accumulated later) and per-expert counters.
// ---------------------------------------------------------------------------
__global__ void init_kernel(float* __restrict__ y, int* __restrict__ cnt) {
  size_t i = (size_t)blockIdx.x * blockDim.x + threadIdx.x;
  size_t base = i * 4;
  if (base < (size_t)N_ * D_) {
    float4 z = {0.f, 0.f, 0.f, 0.f};
    *(float4*)(y + base) = z;
  }
  if (blockIdx.x == 0 && threadIdx.x < E_) cnt[threadIdx.x] = 0;
}

// ---------------------------------------------------------------------------
// Conversion kernels: fp32 -> bf16, weights transposed so WMMA B-loads are
// contiguous along K. W1t[e][h][d] = W1[e][d][h]; W2t[e][d][h] = W2[e][h][d].
// ---------------------------------------------------------------------------
__global__ void conv_x_kernel(const float* __restrict__ x,
                              unsigned short* __restrict__ xbf) {
  size_t i = (size_t)blockIdx.x * blockDim.x + threadIdx.x;
  if (i < (size_t)N_ * D_) xbf[i] = f2bf(x[i]);
}

__global__ void conv_w1_kernel(const float* __restrict__ W1,
                               unsigned short* __restrict__ w1t) {
  size_t i = (size_t)blockIdx.x * blockDim.x + threadIdx.x;
  if (i >= (size_t)E_ * H_ * D_) return;
  int e = (int)(i / (H_ * D_));
  int r = (int)(i % (H_ * D_));
  int h = r / D_, d = r % D_;
  w1t[i] = f2bf(W1[((size_t)e * D_ + d) * H_ + h]);
}

__global__ void conv_w2_kernel(const float* __restrict__ W2,
                               unsigned short* __restrict__ w2t) {
  size_t i = (size_t)blockIdx.x * blockDim.x + threadIdx.x;
  if (i >= (size_t)E_ * D_ * H_) return;
  int e = (int)(i / (D_ * H_));
  int r = (int)(i % (D_ * H_));
  int d = r / H_, h = r % H_;
  w2t[i] = f2bf(W2[((size_t)e * H_ + h) * D_ + d]);
}

// ---------------------------------------------------------------------------
// Kernel 1: Gaussian log-prob gating, top-2 + softmax, per-expert token lists.
// One wave32 per token; fp32 exact math (routing is a discrete decision).
// ---------------------------------------------------------------------------
__global__ __launch_bounds__(256)
void gating_kernel(const float* __restrict__ x, const float* __restrict__ mus,
                   const float* __restrict__ lsg,
                   float* __restrict__ lp_out, float* __restrict__ idx_out,
                   int* __restrict__ cnt, int* __restrict__ tokl,
                   float* __restrict__ wtl) {
  const int lane = threadIdx.x & 31;
  const int wave = threadIdx.x >> 5;
  const int n = blockIdx.x * 8 + wave;
  if (n >= N_) return;
  const int b = n / T_;

  // each lane owns 16 contiguous floats of the token's D=512 features
  float xv[16];
  const float* xp = x + (size_t)n * D_ + lane * 16;
#pragma unroll
  for (int j = 0; j < 16; j += 4) {
    float4 t = *(const float4*)(xp + j);
    xv[j] = t.x; xv[j + 1] = t.y; xv[j + 2] = t.z; xv[j + 3] = t.w;
  }

  float lp[E_];
#pragma unroll
  for (int e = 0; e < E_; e++) {
    const float* mp = mus + ((size_t)b * E_ + e) * D_ + lane * 16;
    const float* sp = lsg + ((size_t)b * E_ + e) * D_ + lane * 16;
    float s = 0.f;
#pragma unroll
    for (int j = 0; j < 16; j += 4) {
      float4 m4 = *(const float4*)(mp + j);
      float4 s4 = *(const float4*)(sp + j);
      float mm[4] = {m4.x, m4.y, m4.z, m4.w};
      float ss[4] = {s4.x, s4.y, s4.z, s4.w};
#pragma unroll
      for (int q = 0; q < 4; q++) {
        float z = (xv[j + q] - mm[q]) * expf(-ss[q]);
        s += -0.5f * z * z - ss[q];
      }
    }
#pragma unroll
    for (int off = 16; off > 0; off >>= 1) s += __shfl_xor(s, off, 32);
    lp[e] = s - 0.5f * LOG2PI_ * (float)D_;
  }

  // top-2 (unrolled, register-only)
  float best = lp[0], sec = -INFINITY;
  int bi = 0, si = 0;
#pragma unroll
  for (int e = 1; e < E_; e++) {
    if (lp[e] > best) { sec = best; si = bi; best = lp[e]; bi = e; }
    else if (lp[e] > sec) { sec = lp[e]; si = e; }
  }
  float w0 = 1.0f / (1.0f + expf(sec - best));
  float w1 = 1.0f - w0;

  if (lane == 0) {
    float* o = lp_out + (size_t)n * E_;
#pragma unroll
    for (int e = 0; e < E_; e++) o[e] = lp[e];
    idx_out[n * 2 + 0] = (float)bi;
    idx_out[n * 2 + 1] = (float)si;
    int p0 = atomicAdd(&cnt[bi], 1);
    tokl[bi * N_ + p0] = n; wtl[bi * N_ + p0] = w0;
    int p1 = atomicAdd(&cnt[si], 1);
    tokl[si * N_ + p1] = n; wtl[si * N_ + p1] = w1;
  }
}

// ---------------------------------------------------------------------------
// Kernel 2: fused expert MLP over gathered 32-token tiles.
// grid = (N/32 tiles, E experts), 256 threads = 8 waves.
// Each B tile (32x16 bf16, 1 KB) held in registers feeds TWO wmma ops
// (row tiles 0 and 1) -> 32 FLOP/byte from L2 instead of 16.
// GEMM1 (K=512) -> bias + exact GELU -> GEMM2 (K=1024) -> gated atomic scatter.
// ---------------------------------------------------------------------------
__global__ __launch_bounds__(256)
void moe_kernel(const unsigned short* __restrict__ xbf,
                const unsigned short* __restrict__ w1t,
                const unsigned short* __restrict__ w2t,
                const float* __restrict__ b1, const float* __restrict__ b2,
                const int* __restrict__ cnt, const int* __restrict__ tokl,
                const float* __restrict__ wtl, float* __restrict__ y) {
  __shared__ __align__(16) unsigned short x_lds[TM_ * 520];   // pad 8 halves
  __shared__ __align__(16) unsigned short h_lds[TM_ * 1032];  // pad 8 halves
  __shared__ int tok_s[TM_];
  __shared__ float wt_s[TM_];

  const int e = blockIdx.y;
  const int count = cnt[e];
  const int base = blockIdx.x * TM_;
  if (base >= count) return;
  const int nvalid = (count - base < TM_) ? (count - base) : TM_;

  const int tid = threadIdx.x;
  const int lane = tid & 31, wave = tid >> 5;
  const int m = lane & 15, half = lane >> 4;

  if (tid < TM_) {
    if (tid < nvalid) {
      tok_s[tid] = tokl[e * N_ + base + tid];
      wt_s[tid] = wtl[e * N_ + base + tid];
    } else {
      tok_s[tid] = 0; wt_s[tid] = 0.f;
    }
  }
  __syncthreads();

  // gather TM_ token rows (bf16) into LDS
  for (int c = tid; c < TM_ * 64; c += 256) {
    int row = c >> 6, off = (c & 63) << 3;
    ushort8 v = {};
    if (row < nvalid) v = *(const ushort8*)(xbf + (size_t)tok_s[row] * D_ + off);
    *(ushort8*)(x_lds + row * 520 + off) = v;
  }
  __syncthreads();

  // ---- GEMM1: h[32,1024] = x_tile @ W1[e]; each wave owns 128 columns ----
  {
    v8f acc0[8] = {};
    v8f acc1[8] = {};
    const int n0w = wave * 128;
    for (int k0 = 0; k0 < D_; k0 += 32) {
      // A 16x32 bf16 layout: lanes 0-15 K=0-7/16-23, lanes 16-31 K=8-15/24-31
      BF16x16 a0, a1;
      a0.u[0] = *(const ushort8*)(x_lds + m * 520 + k0 + half * 8);
      a0.u[1] = *(const ushort8*)(x_lds + m * 520 + k0 + 16 + half * 8);
      a1.u[0] = *(const ushort8*)(x_lds + (16 + m) * 520 + k0 + half * 8);
      a1.u[1] = *(const ushort8*)(x_lds + (16 + m) * 520 + k0 + 16 + half * 8);
#pragma unroll
      for (int c = 0; c < 8; c++) {
        const unsigned short* bp =
            w1t + ((size_t)(e * H_ + n0w + c * 16 + m)) * D_ + k0;
        if (k0 + 32 < D_) __builtin_prefetch(bp + 32, 0, 1);
        BF16x16 bb;
        bb.u[0] = *(const ushort8*)(bp + half * 8);
        bb.u[1] = *(const ushort8*)(bp + 16 + half * 8);
        acc0[c] = __builtin_amdgcn_wmma_f32_16x16x32_bf16(
            false, a0.v, false, bb.v, (short)0, acc0[c], false, false);
        acc1[c] = __builtin_amdgcn_wmma_f32_16x16x32_bf16(
            false, a1.v, false, bb.v, (short)0, acc1[c], false, false);
      }
    }
#pragma unroll
    for (int c = 0; c < 8; c++) {
      int col = n0w + c * 16 + m;
      float bias = b1[e * H_ + col];
#pragma unroll
      for (int r = 0; r < 8; r++) {
        float v0 = acc0[c][r] + bias;                        // row = r + half*8
        v0 = 0.5f * v0 * (1.0f + erff(v0 * 0.70710678118654752f));
        h_lds[(r + half * 8) * 1032 + col] = f2bf(v0);
        float v1 = acc1[c][r] + bias;                        // row = 16 + r + half*8
        v1 = 0.5f * v1 * (1.0f + erff(v1 * 0.70710678118654752f));
        h_lds[(16 + r + half * 8) * 1032 + col] = f2bf(v1);
      }
    }
  }
  __syncthreads();

  // ---- GEMM2: out[32,512] = h @ W2[e]; each wave owns 64 columns ----
  {
    v8f acc0[4] = {};
    v8f acc1[4] = {};
    const int n0w = wave * 64;
    for (int k0 = 0; k0 < H_; k0 += 32) {
      BF16x16 a0, a1;
      a0.u[0] = *(const ushort8*)(h_lds + m * 1032 + k0 + half * 8);
      a0.u[1] = *(const ushort8*)(h_lds + m * 1032 + k0 + 16 + half * 8);
      a1.u[0] = *(const ushort8*)(h_lds + (16 + m) * 1032 + k0 + half * 8);
      a1.u[1] = *(const ushort8*)(h_lds + (16 + m) * 1032 + k0 + 16 + half * 8);
#pragma unroll
      for (int c = 0; c < 4; c++) {
        const unsigned short* bp =
            w2t + ((size_t)(e * D_ + n0w + c * 16 + m)) * H_ + k0;
        if (k0 + 32 < H_) __builtin_prefetch(bp + 32, 0, 1);
        BF16x16 bb;
        bb.u[0] = *(const ushort8*)(bp + half * 8);
        bb.u[1] = *(const ushort8*)(bp + 16 + half * 8);
        acc0[c] = __builtin_amdgcn_wmma_f32_16x16x32_bf16(
            false, a0.v, false, bb.v, (short)0, acc0[c], false, false);
        acc1[c] = __builtin_amdgcn_wmma_f32_16x16x32_bf16(
            false, a1.v, false, bb.v, (short)0, acc1[c], false, false);
      }
    }
#pragma unroll
    for (int c = 0; c < 4; c++) {
      int col = n0w + c * 16 + m;
      float bias = b2[e * D_ + col];
#pragma unroll
      for (int r = 0; r < 8; r++) {
        int row0 = r + half * 8;
        int row1 = 16 + r + half * 8;
        if (row0 < nvalid) {
          float v = acc0[c][r] + bias;
          atomicAdd(&y[(size_t)tok_s[row0] * D_ + col], wt_s[row0] * v);
        }
        if (row1 < nvalid) {
          float v = acc1[c][r] + bias;
          atomicAdd(&y[(size_t)tok_s[row1] * D_ + col], wt_s[row1] * v);
        }
      }
    }
  }
}

// ---------------------------------------------------------------------------
// Host launcher.
// Inputs: x, mus, log_sigmas, W1, b1, W2, b2, top_k(=2, hardcoded)
// Output (concatenated): y[N*D], log_probs[N*E], top_idx[N*2] (as float)
// ---------------------------------------------------------------------------
extern "C" void kernel_launch(void* const* d_in, const int* in_sizes, int n_in,
                              void* d_out, int out_size, void* d_ws,
                              size_t ws_size, hipStream_t stream) {
  (void)in_sizes; (void)n_in; (void)out_size; (void)ws_size;

  const float* x   = (const float*)d_in[0];
  const float* mus = (const float*)d_in[1];
  const float* lsg = (const float*)d_in[2];
  const float* W1  = (const float*)d_in[3];
  const float* b1  = (const float*)d_in[4];
  const float* W2  = (const float*)d_in[5];
  const float* b2  = (const float*)d_in[6];

  float* y       = (float*)d_out;
  float* lp_out  = y + (size_t)N_ * D_;
  float* idx_out = lp_out + (size_t)N_ * E_;

  // workspace layout (~25 MB): counters | x_bf16 | W1t_bf16 | W2t_bf16 | lists
  char* w = (char*)d_ws;
  int* cnt            = (int*)w;
  unsigned short* xbf = (unsigned short*)(w + 256);
  unsigned short* w1t = xbf + (size_t)N_ * D_;
  unsigned short* w2t = w1t + (size_t)E_ * H_ * D_;
  int* tokl           = (int*)(w2t + (size_t)E_ * D_ * H_);
  float* wtl          = (float*)(tokl + (size_t)E_ * N_);

  const size_t nd = (size_t)N_ * D_;
  const size_t nw = (size_t)E_ * H_ * D_;

  init_kernel<<<(unsigned)((nd / 4 + 255) / 256), 256, 0, stream>>>(y, cnt);
  conv_x_kernel<<<(unsigned)((nd + 255) / 256), 256, 0, stream>>>(x, xbf);
  conv_w1_kernel<<<(unsigned)((nw + 255) / 256), 256, 0, stream>>>(W1, w1t);
  conv_w2_kernel<<<(unsigned)((nw + 255) / 256), 256, 0, stream>>>(W2, w2t);
  gating_kernel<<<N_ / 8, 256, 0, stream>>>(x, mus, lsg, lp_out, idx_out, cnt,
                                            tokl, wtl);
  moe_kernel<<<dim3(N_ / TM_, E_), 256, 0, stream>>>(xbf, w1t, w2t, b1, b2,
                                                     cnt, tokl, wtl, y);
}
